// GATv2_Classifier_49813030699381
// MI455X (gfx1250) — matrix operand order repaired and checked by reading, not compiled, not run on hardware
//
#include <hip/hip_runtime.h>
#include <hip/hip_bf16.h>

typedef __attribute__((ext_vector_type(16))) _Float16 v16h;
typedef __attribute__((ext_vector_type(8)))  _Float16 v8h;
typedef __attribute__((ext_vector_type(8)))  float    v8f;

#define NEG_SLOPE 0.2f

// ---------- helpers: order-preserving f32 <-> u32 for atomic max ----------
__device__ __forceinline__ unsigned enc_f32(float f) {
    unsigned b = __float_as_uint(f);
    return (b & 0x80000000u) ? ~b : (b | 0x80000000u);
}
__device__ __forceinline__ float dec_f32(unsigned u) {
    unsigned b = (u & 0x80000000u) ? (u & 0x7FFFFFFFu) : ~u;
    return __uint_as_float(b);
}
__device__ __forceinline__ float lrelu(float v) {
    return v > 0.0f ? v : NEG_SLOPE * v;
}

// ---------- small utility kernels ----------
__global__ void k_build_edges(const int* __restrict__ ei, int* __restrict__ src,
                              int* __restrict__ dst, int E, int N) {
    int t = blockIdx.x * blockDim.x + threadIdx.x;
    int EP = E + N;
    if (t >= EP) return;
    if (t < E) { src[t] = ei[t]; dst[t] = ei[E + t]; }
    else       { src[t] = t - E; dst[t] = t - E; }
}

__global__ void k_f32_to_f16(const float* __restrict__ in, _Float16* __restrict__ out, long n) {
    long i = (long)blockIdx.x * blockDim.x + threadIdx.x;
    if (i < n) out[i] = (_Float16)in[i];
}

// W [K, Nsrc] row-major (f32) -> Wt [Npad, K] (f16), zero-padded columns
__global__ void k_transpose_f16(const float* __restrict__ W, _Float16* __restrict__ Wt,
                                int K, int Nsrc, int Npad) {
    int t = blockIdx.x * blockDim.x + threadIdx.x;
    if (t >= Npad * K) return;
    int n = t / K, k = t - n * K;
    float v = (n < Nsrc) ? W[(size_t)k * Nsrc + n] : 0.0f;
    Wt[t] = (_Float16)v;
}

__global__ void k_fill_u32(unsigned* __restrict__ p, unsigned v, long n) {
    long i = (long)blockIdx.x * blockDim.x + threadIdx.x;
    if (i < n) p[i] = v;
}

// ---------- WMMA GEMM: C[M,ldc] = A[M,K] * Bt[Nmat,K]^T (+bias), f16 in, f32 out ----------
// One wave32 computes a 16x(16*NT) strip: NT accumulators share each A load.
// Layouts per CDNA5 ISA 7.12.2:
//   A (16-bit 16x32): lanes 0-15 row M=l, K {0..7,16..23}; lanes 16-31 K {8..15,24..31}
//   B (16-bit 32x16): lanes 0-15 col N=l, K {0..7,16..23}; lanes 16-31 K {8..15,24..31}
//   C/D f32: VGPR r -> M = (lane<16 ? r : 8+r), N = lane&15
template<int NT>
__global__ void k_gemm_wmma(const _Float16* __restrict__ A, const _Float16* __restrict__ Bt,
                            float* __restrict__ C, int M, int K, int Nmat,
                            int ldc, int nstore, const float* __restrict__ bias) {
    int tid  = blockIdx.x * blockDim.x + threadIdx.x;
    int wave = tid >> 5;
    int lane = tid & 31;
    int groupsN = (Nmat >> 4) / NT;
    int tilesM  = M >> 4;
    int tm = wave / groupsN;
    int tg = wave - tm * groupsN;
    if (tm >= tilesM) return;                 // wave-uniform exit (EXEC all-1 inside wave)

    int g = lane >> 4;                        // half-wave group
    int l = lane & 15;
    const _Float16* pa = A + (size_t)(tm * 16 + l) * K + g * 8;
    const _Float16* pb[NT];
    #pragma unroll
    for (int j = 0; j < NT; ++j)
        pb[j] = Bt + (size_t)((tg * NT + j) * 16 + l) * K + g * 8;

    v8f acc[NT] = {};
    for (int kk = 0; kk < K; kk += 32) {
        union { v16h v; v8h h[2]; } ua;
        ua.h[0] = *(const v8h*)(pa + kk);          // K = g*8 .. g*8+7
        ua.h[1] = *(const v8h*)(pa + kk + 16);     // K = g*8+16 .. g*8+23
        __builtin_prefetch(pa + kk + 32, 0, 3);    // next A block, WGP scope, speculative
        #pragma unroll
        for (int j = 0; j < NT; ++j) {
            union { v16h v; v8h h[2]; } ub;
            ub.h[0] = *(const v8h*)(pb[j] + kk);       // K = g*8 .. g*8+7
            ub.h[1] = *(const v8h*)(pb[j] + kk + 16);  // K = g*8+16 .. g*8+23
            acc[j] = __builtin_amdgcn_wmma_f32_16x16x32_f16(
                         false, ua.v, false, ub.v, (short)0, acc[j], false, false);
        }
    }

    #pragma unroll
    for (int j = 0; j < NT; ++j) {
        int col = (tg * NT + j) * 16 + l;
        if (col < nstore) {
            float badd = bias ? bias[col] : 0.0f;
            #pragma unroll
            for (int r = 0; r < 8; ++r) {
                int row = tm * 16 + g * 8 + r;
                C[(size_t)row * ldc + col] = acc[j][r] + badd;
            }
        }
    }
}

// ---------- GATv2 edge kernels (C multiple of 4, float4 vectorized) ----------
// score[e,h] = sum_c att[h,c] * leakyrelu(xl[src,h,c] + xr[dst,h,c]); atomic max into mmax[dst,h]
__global__ void k_edge_score(const float* __restrict__ xl, const float* __restrict__ xr,
                             const int* __restrict__ src, const int* __restrict__ dst,
                             const float* __restrict__ att, float* __restrict__ score,
                             unsigned* __restrict__ mmax, int EP, int Hh, int C) {
    int t = blockIdx.x * blockDim.x + threadIdx.x;
    if (t >= EP * Hh) return;
    int e = t / Hh, h = t - e * Hh;
    int s = src[e], d = dst[e];
    const float4* pl = (const float4*)(xl  + (size_t)s * Hh * C + (size_t)h * C);
    const float4* pr = (const float4*)(xr  + (size_t)d * Hh * C + (size_t)h * C);
    const float4* pa = (const float4*)(att + (size_t)h * C);
    float acc = 0.0f;
    int C4 = C >> 2;
    for (int c = 0; c < C4; ++c) {
        float4 a = pl[c], b = pr[c], w = pa[c];
        acc += lrelu(a.x + b.x) * w.x;
        acc += lrelu(a.y + b.y) * w.y;
        acc += lrelu(a.z + b.z) * w.z;
        acc += lrelu(a.w + b.w) * w.w;
    }
    score[t] = acc;
    atomicMax(mmax + (size_t)d * Hh + h, enc_f32(acc));
}

// a = exp(score - m[dst]); score <- a; atomic add into ssum[dst,h]
__global__ void k_edge_exp(float* __restrict__ score, const unsigned* __restrict__ mmax,
                           const int* __restrict__ dst, float* __restrict__ ssum,
                           int EP, int Hh) {
    int t = blockIdx.x * blockDim.x + threadIdx.x;
    if (t >= EP * Hh) return;
    int e = t / Hh, h = t - e * Hh;
    int d = dst[e];
    float m = dec_f32(mmax[(size_t)d * Hh + h]);
    float a = expf(score[t] - m);
    score[t] = a;
    atomicAdd(ssum + (size_t)d * Hh + h, a);
}

// agg[dst,h,:] += (a / ssum[dst,h]) * xl[src,h,:]
__global__ void k_edge_agg(const float* __restrict__ score, const float* __restrict__ ssum,
                           const float* __restrict__ xl, const int* __restrict__ src,
                           const int* __restrict__ dst, float* __restrict__ agg,
                           int EP, int Hh, int C) {
    int t = blockIdx.x * blockDim.x + threadIdx.x;
    if (t >= EP * Hh) return;
    int e = t / Hh, h = t - e * Hh;
    int s = src[e], d = dst[e];
    float alpha = score[t] / ssum[(size_t)d * Hh + h];
    const float4* pl = (const float4*)(xl + (size_t)s * Hh * C + (size_t)h * C);
    float*        po = agg + (size_t)d * Hh * C + (size_t)h * C;
    int C4 = C >> 2;
    for (int c = 0; c < C4; ++c) {
        float4 a = pl[c];
        atomicAdd(po + 4 * c + 0, alpha * a.x);
        atomicAdd(po + 4 * c + 1, alpha * a.y);
        atomicAdd(po + 4 * c + 2, alpha * a.z);
        atomicAdd(po + 4 * c + 3, alpha * a.w);
    }
}

// h = relu(agg + bias) -> f16 (feeds next GEMM)
__global__ void k_finalize(const float* __restrict__ agg, const float* __restrict__ bias,
                           _Float16* __restrict__ hout, long n, int HC) {
    long i = (long)blockIdx.x * blockDim.x + threadIdx.x;
    if (i >= n) return;
    float v = agg[i] + bias[i % HC];
    v = v > 0.0f ? v : 0.0f;
    hout[i] = (_Float16)v;
}

// ---------- host-side orchestration ----------
static inline int nblk(long n) { return (int)((n + 255) / 256); }

extern "C" void kernel_launch(void* const* d_in, const int* in_sizes, int n_in,
                              void* d_out, int out_size, void* d_ws, size_t ws_size,
                              hipStream_t stream) {
    (void)in_sizes; (void)n_in; (void)out_size; (void)ws_size;
    const float* x    = (const float*)d_in[0];
    const int*   ei   = (const int*)  d_in[1];
    const float* Wl1  = (const float*)d_in[2];
    const float* Wr1  = (const float*)d_in[3];
    const float* att1 = (const float*)d_in[4];
    const float* b1   = (const float*)d_in[5];
    const float* Wl2  = (const float*)d_in[6];
    const float* Wr2  = (const float*)d_in[7];
    const float* att2 = (const float*)d_in[8];
    const float* b2   = (const float*)d_in[9];
    const float* Wlin = (const float*)d_in[10];
    const float* blin = (const float*)d_in[11];
    float* out = (float*)d_out;

    const int N = 50000, E = 800000, IN = 256, OUT = 64, H = 4, NC = 10;
    const int EP  = E + N;        // 850000 edges incl. self-loops
    const int HC1 = H * OUT;      // 256

    // deterministic scratch carving (256B aligned)
    char* p = (char*)d_ws;
    auto alloc = [&](size_t bytes) -> void* {
        void* r = (void*)p;
        p += (bytes + 255) & ~(size_t)255;
        return r;
    };
    int*      src   = (int*)alloc((size_t)EP * 4);
    int*      dst   = (int*)alloc((size_t)EP * 4);
    _Float16* Wl1t  = (_Float16*)alloc((size_t)256 * 256 * 2);
    _Float16* Wr1t  = (_Float16*)alloc((size_t)256 * 256 * 2);
    _Float16* Wl2t  = (_Float16*)alloc((size_t)64 * 256 * 2);
    _Float16* Wr2t  = (_Float16*)alloc((size_t)64 * 256 * 2);
    _Float16* Wlint = (_Float16*)alloc((size_t)16 * 64 * 2);
    _Float16* x16   = (_Float16*)alloc((size_t)N * IN * 2);
    float*    xl1   = (float*)alloc((size_t)N * HC1 * 4);
    float*    xr1   = (float*)alloc((size_t)N * HC1 * 4);
    float*    sc1   = (float*)alloc((size_t)EP * H * 4);
    unsigned* mm1   = (unsigned*)alloc((size_t)N * H * 4);
    float*    ss1   = (float*)alloc((size_t)N * H * 4);
    float*    agg1  = (float*)alloc((size_t)N * HC1 * 4);
    _Float16* h16   = (_Float16*)alloc((size_t)N * HC1 * 2);
    float*    xl2   = (float*)alloc((size_t)N * OUT * 4);
    float*    xr2   = (float*)alloc((size_t)N * OUT * 4);
    float*    sc2   = (float*)alloc((size_t)EP * 4);
    unsigned* mm2   = (unsigned*)alloc((size_t)N * 4);
    float*    ss2   = (float*)alloc((size_t)N * 4);
    float*    agg2  = (float*)alloc((size_t)N * OUT * 4);
    _Float16* h216  = (_Float16*)alloc((size_t)N * OUT * 2);

    dim3 B256(256);

    // edges + f16 conversions / weight transposes
    k_build_edges<<<nblk(EP), B256, 0, stream>>>(ei, src, dst, E, N);
    k_f32_to_f16<<<nblk((long)N * IN), B256, 0, stream>>>(x, x16, (long)N * IN);
    k_transpose_f16<<<nblk(256 * 256), B256, 0, stream>>>(Wl1, Wl1t, 256, 256, 256);
    k_transpose_f16<<<nblk(256 * 256), B256, 0, stream>>>(Wr1, Wr1t, 256, 256, 256);
    k_transpose_f16<<<nblk(64 * 256), B256, 0, stream>>>(Wl2, Wl2t, 256, 64, 64);
    k_transpose_f16<<<nblk(64 * 256), B256, 0, stream>>>(Wr2, Wr2t, 256, 64, 64);
    k_transpose_f16<<<nblk(16 * 64), B256, 0, stream>>>(Wlin, Wlint, 64, 10, 16);

    // ---- layer 1: xl1 = x@Wl1, xr1 = x@Wr1 (WMMA, 4 N-tiles per wave) ----
    {
        long waves = (long)(N / 16) * (256 / 64);     // groupsN = 256/(16*4) = 4
        int  blocks = nblk(waves * 32);
        k_gemm_wmma<4><<<blocks, B256, 0, stream>>>(x16, Wl1t, xl1, N, 256, 256, 256, 256, nullptr);
        k_gemm_wmma<4><<<blocks, B256, 0, stream>>>(x16, Wr1t, xr1, N, 256, 256, 256, 256, nullptr);
    }
    k_fill_u32<<<nblk((long)N * H), B256, 0, stream>>>(mm1, 0u, (long)N * H);
    k_fill_u32<<<nblk((long)N * H), B256, 0, stream>>>((unsigned*)ss1, 0u, (long)N * H);
    k_fill_u32<<<nblk((long)N * HC1), B256, 0, stream>>>((unsigned*)agg1, 0u, (long)N * HC1);
    k_edge_score<<<nblk((long)EP * H), B256, 0, stream>>>(xl1, xr1, src, dst, att1, sc1, mm1, EP, H, OUT);
    k_edge_exp<<<nblk((long)EP * H), B256, 0, stream>>>(sc1, mm1, dst, ss1, EP, H);
    k_edge_agg<<<nblk((long)EP * H), B256, 0, stream>>>(sc1, ss1, xl1, src, dst, agg1, EP, H, OUT);
    k_finalize<<<nblk((long)N * HC1), B256, 0, stream>>>(agg1, b1, h16, (long)N * HC1, HC1);

    // ---- layer 2: xl2 = h@Wl2, xr2 = h@Wr2 (WMMA), H=1 ----
    {
        long waves = (long)(N / 16);                  // groupsN = 64/(16*4) = 1
        int  blocks = nblk(waves * 32);
        k_gemm_wmma<4><<<blocks, B256, 0, stream>>>(h16, Wl2t, xl2, N, 256, 64, 64, 64, nullptr);
        k_gemm_wmma<4><<<blocks, B256, 0, stream>>>(h16, Wr2t, xr2, N, 256, 64, 64, 64, nullptr);
    }
    k_fill_u32<<<nblk((long)N), B256, 0, stream>>>(mm2, 0u, (long)N);
    k_fill_u32<<<nblk((long)N), B256, 0, stream>>>((unsigned*)ss2, 0u, (long)N);
    k_fill_u32<<<nblk((long)N * OUT), B256, 0, stream>>>((unsigned*)agg2, 0u, (long)N * OUT);
    k_edge_score<<<nblk((long)EP), B256, 0, stream>>>(xl2, xr2, src, dst, att2, sc2, mm2, EP, 1, OUT);
    k_edge_exp<<<nblk((long)EP), B256, 0, stream>>>(sc2, mm2, dst, ss2, EP, 1);
    k_edge_agg<<<nblk((long)EP), B256, 0, stream>>>(sc2, ss2, xl2, src, dst, agg2, EP, 1, OUT);
    k_finalize<<<nblk((long)N * OUT), B256, 0, stream>>>(agg2, b2, h216, (long)N * OUT, OUT);

    // ---- classifier: out = h2 @ Wlin + blin (WMMA, N padded 10->16, masked store) ----
    {
        long waves = (long)(N / 16);
        int  blocks = nblk(waves * 32);
        k_gemm_wmma<1><<<blocks, B256, 0, stream>>>(h216, Wlint, out, N, 64, 16, NC, NC, blin);
    }
}
